// MultiHeadAttention_7834020348049
// MI455X (gfx1250) — matrix operand order repaired and checked by reading, not compiled, not run on hardware
//
#include <hip/hip_runtime.h>

// ---------------------------------------------------------------------------
// Types for WMMA fragments (gfx1250, wave32)
// ---------------------------------------------------------------------------
typedef __attribute__((ext_vector_type(8)))  __bf16 v8bf;
typedef __attribute__((ext_vector_type(16))) __bf16 v16bf;
typedef __attribute__((ext_vector_type(8)))  float  v8f;

__device__ __forceinline__ unsigned short f32_to_bf16(float f) {
    unsigned int u = __float_as_uint(f);
    u += 0x7FFFu + ((u >> 16) & 1u);   // round-to-nearest-even
    return (unsigned short)(u >> 16);
}
__device__ __forceinline__ float bf16_to_f32(unsigned short h) {
    return __uint_as_float(((unsigned int)h) << 16);
}
__device__ __forceinline__ unsigned int pack_bf16x2(float a, float b) {
    return (unsigned int)f32_to_bf16(a) | ((unsigned int)f32_to_bf16(b) << 16);
}
__device__ __forceinline__ v16bf cat8(v8bf lo, v8bf hi) {
    return __builtin_shufflevector(lo, hi, 0,1,2,3,4,5,6,7,8,9,10,11,12,13,14,15);
}

// ---------------------------------------------------------------------------
// CDNA5 async global->LDS copy (no VGPR staging; tracked by ASYNCcnt).
//   GV addressing: global_load_async_to_lds_b128 vdst(LDS byte offs), v[addr], off
//   Truncating a generic pointer to a __shared__ object yields the LDS offset
//   (LDS aperture: addr[31:0] == wave-relative LDS byte address).
// ---------------------------------------------------------------------------
__device__ __forceinline__ void async_copy_b128(void* lds_dst, const void* gsrc) {
    unsigned lds_off = (unsigned)(unsigned long long)lds_dst;
    asm volatile("global_load_async_to_lds_b128 %0, %1, off"
                 :: "v"(lds_off), "v"(gsrc)
                 : "memory");
}
__device__ __forceinline__ void wait_async0() {
    asm volatile("s_wait_asynccnt 0" ::: "memory");
}

// ---------------------------------------------------------------------------
// Problem constants
// ---------------------------------------------------------------------------
#define DIMX     1024
#define HEADS    8
#define DIM_HEAD 128
#define N_ROWS   (16 * 2048)        // bs * patch = 32768
#define ATT_SCALE 0.088388347648318447f   // 128^-0.5

// ---------------------------------------------------------------------------
// Kernel 0: f32 -> bf16 bulk convert (weights)
// ---------------------------------------------------------------------------
__global__ __launch_bounds__(256)
void k_f32_to_bf16(const float* __restrict__ in, unsigned short* __restrict__ out, int n) {
    int i = (blockIdx.x * 256 + threadIdx.x) * 4;
    if (i >= n) return;
    float4 v = *(const float4*)(in + i);
    unsigned int* o = (unsigned int*)(out + i);
    o[0] = pack_bf16x2(v.x, v.y);
    o[1] = pack_bf16x2(v.z, v.w);
}

// ---------------------------------------------------------------------------
// Kernel 1: LayerNorm over 1024-wide rows, one wave32 per row, bf16 output
// ---------------------------------------------------------------------------
__global__ __launch_bounds__(256)
void k_layernorm_bf16(const float* __restrict__ x,
                      const float* __restrict__ gamma,
                      const float* __restrict__ beta,
                      unsigned short* __restrict__ out) {
    const int wave = threadIdx.x >> 5;
    const int lane = threadIdx.x & 31;
    const size_t row = (size_t)blockIdx.x * 8 + wave;
    const float* xr = x + row * DIMX;

    float4 v[8];
    float s = 0.f, s2 = 0.f;
    #pragma unroll
    for (int i = 0; i < 8; ++i) {
        v[i] = *(const float4*)(xr + i * 128 + lane * 4);
        s  += v[i].x + v[i].y + v[i].z + v[i].w;
        s2 += v[i].x * v[i].x + v[i].y * v[i].y + v[i].z * v[i].z + v[i].w * v[i].w;
    }
    #pragma unroll
    for (int off = 16; off > 0; off >>= 1) {
        s  += __shfl_xor(s,  off);
        s2 += __shfl_xor(s2, off);
    }
    const float mu  = s * (1.f / DIMX);
    const float var = s2 * (1.f / DIMX) - mu * mu;
    const float inv = rsqrtf(var + 1e-5f);

    unsigned int* o32 = (unsigned int*)(out + row * DIMX);
    #pragma unroll
    for (int i = 0; i < 8; ++i) {
        int j = i * 128 + lane * 4;
        float4 g = *(const float4*)(gamma + j);
        float4 b = *(const float4*)(beta + j);
        float y0 = (v[i].x - mu) * inv * g.x + b.x;
        float y1 = (v[i].y - mu) * inv * g.y + b.y;
        float y2 = (v[i].z - mu) * inv * g.z + b.z;
        float y3 = (v[i].w - mu) * inv * g.w + b.w;
        o32[j / 2]     = pack_bf16x2(y0, y1);
        o32[j / 2 + 1] = pack_bf16x2(y2, y3);
    }
}

// ---------------------------------------------------------------------------
// Kernel 2: bf16 WMMA GEMM  C[M,N] = A[M,K] * B[N,K]^T  (B = weight, row-major)
//   Tile: BM=128, BN=128, BK=32; 256 threads = 8 waves in 2(m) x 4(n) grid.
//   Each wave: 4 m-subtiles x 2 n-subtiles of v_wmma_f32_16x16x32_bf16.
//   Double-buffered LDS, filled by GLOBAL_LOAD_ASYNC_TO_LDS_B128 (ASYNCcnt),
//   so next-tile DMA overlaps this tile's WMMA work.
// ---------------------------------------------------------------------------
#define BM 128
#define BN 128
#define BK 32
#define LDSS 40   // 32 + 8 pad (ushort elements); row stride 80B (16B aligned)

template <bool OUT_BF16>
__global__ __launch_bounds__(256)
void k_wmma_gemm_bf16(const unsigned short* __restrict__ A,  // [M,K]
                      const unsigned short* __restrict__ B,  // [N,K]
                      void* __restrict__ C,                  // [M,N]
                      int M, int N, int K) {
    __shared__ unsigned short Asm[2][BM * LDSS];
    __shared__ unsigned short Bsm[2][BN * LDSS];

    const int tid  = threadIdx.x;
    const int lane = tid & 31;
    const int wave = tid >> 5;
    const int wm   = wave >> 2;          // 0..1  -> 64 rows
    const int wn   = wave & 3;           // 0..3  -> 32 cols
    const int m0   = blockIdx.y * BM;
    const int n0   = blockIdx.x * BN;

    v8f acc[4][2];
    #pragma unroll
    for (int mi = 0; mi < 4; ++mi)
        #pragma unroll
        for (int ni = 0; ni < 2; ++ni)
            acc[mi][ni] = (v8f){0.f, 0.f, 0.f, 0.f, 0.f, 0.f, 0.f, 0.f};

    // Issue async global->LDS copies for K-stage kt into buffer buf.
    // 512 16B chunks per tile; 2 per thread per matrix.
    auto issue_tile = [&](int kt, int buf) {
        #pragma unroll
        for (int i = 0; i < 2; ++i) {
            int c   = tid + i * 256;
            int row = c >> 2;
            int kc  = c & 3;
            async_copy_b128(&Asm[buf][row * LDSS + kc * 8],
                            A + (size_t)(m0 + row) * K + (size_t)kt * BK + kc * 8);
            async_copy_b128(&Bsm[buf][row * LDSS + kc * 8],
                            B + (size_t)(n0 + row) * K + (size_t)kt * BK + kc * 8);
        }
    };

    const int NT = K / BK;
    issue_tile(0, 0);
    wait_async0();
    __syncthreads();

    const int r = lane & 15;       // row (A) / column (B) within 16-tile
    const int h = lane >> 4;       // lane half selects K sub-range

    for (int kt = 0; kt < NT; ++kt) {
        int buf = kt & 1;
        if (kt + 1 < NT) issue_tile(kt + 1, buf ^ 1);   // DMA overlaps compute

        // A fragments: lane half h -> k chunks {8h..8h+7} and {16+8h..23+8h}
        v16bf afrag[4];
        #pragma unroll
        for (int mi = 0; mi < 4; ++mi) {
            int row = wm * 64 + mi * 16 + r;
            v8bf lo = *(const v8bf*)&Asm[buf][row * LDSS + 8 * h];
            v8bf hi = *(const v8bf*)&Asm[buf][row * LDSS + 16 + 8 * h];
            afrag[mi] = cat8(lo, hi);
        }
        // B fragments: lane = column, half h -> k chunks {8h..8h+7} and {16+8h..23+8h}
        v16bf bfrag[2];
        #pragma unroll
        for (int ni = 0; ni < 2; ++ni) {
            int row = wn * 32 + ni * 16 + r;
            v8bf lo = *(const v8bf*)&Bsm[buf][row * LDSS + 8 * h];
            v8bf hi = *(const v8bf*)&Bsm[buf][row * LDSS + 16 + 8 * h];
            bfrag[ni] = cat8(lo, hi);
        }
        #pragma unroll
        for (int mi = 0; mi < 4; ++mi)
            #pragma unroll
            for (int ni = 0; ni < 2; ++ni)
                acc[mi][ni] = __builtin_amdgcn_wmma_f32_16x16x32_bf16(
                    false, afrag[mi], false, bfrag[ni],
                    (short)0, acc[mi][ni], false, false);

        wait_async0();       // my next-stage DMA chunks have landed in LDS
        __syncthreads();     // everyone's have
    }

    // Epilogue: C layout — VGPR v, lane half h -> M = 16*mi + v + 8h, N = lane&15
    const int nl = lane & 15;
    #pragma unroll
    for (int mi = 0; mi < 4; ++mi) {
        #pragma unroll
        for (int ni = 0; ni < 2; ++ni) {
            int mbase = m0 + wm * 64 + mi * 16 + h * 8;
            int ncol  = n0 + wn * 32 + ni * 16 + nl;
            #pragma unroll
            for (int v = 0; v < 8; ++v) {
                float val = acc[mi][ni][v];
                size_t idx = (size_t)(mbase + v) * N + ncol;
                if (OUT_BF16) ((unsigned short*)C)[idx] = f32_to_bf16(val);
                else          ((float*)C)[idx] = val;
            }
        }
    }
}

// ---------------------------------------------------------------------------
// Kernel 3: per-row 8x8 attention over heads (K=128), one wave per row.
//   Q/K/V rows staged into LDS via async copies; softmax over the 4-lane
//   head group via shuffles.
// ---------------------------------------------------------------------------
__global__ __launch_bounds__(256)
void k_head_attention(const unsigned short* __restrict__ Qp,   // [rows, 1024]
                      const unsigned short* __restrict__ KVp,  // [rows, 2048] (K | V)
                      unsigned short* __restrict__ Ctx) {      // [rows, 1024]
    __shared__ unsigned short sQ[8][DIMX];
    __shared__ unsigned short sKV[8][2 * DIMX];

    const int wave = threadIdx.x >> 5;
    const int lane = threadIdx.x & 31;
    const size_t row = (size_t)blockIdx.x * 8 + wave;

    #pragma unroll
    for (int i = 0; i < 4; ++i)
        async_copy_b128(&sQ[wave][i * 256 + lane * 8],
                        Qp + row * DIMX + i * 256 + lane * 8);
    #pragma unroll
    for (int i = 0; i < 8; ++i)
        async_copy_b128(&sKV[wave][i * 256 + lane * 8],
                        KVp + row * 2 * DIMX + i * 256 + lane * 8);
    wait_async0();
    __syncthreads();

    const int h  = lane >> 2;
    const int g0 = (lane & 3) * 2, g1 = g0 + 1;

    float s0 = 0.f, s1 = 0.f;
    for (int d = 0; d < DIM_HEAD; ++d) {
        float qf = bf16_to_f32(sQ[wave][h * DIM_HEAD + d]);
        s0 += qf * bf16_to_f32(sKV[wave][g0 * DIM_HEAD + d]);
        s1 += qf * bf16_to_f32(sKV[wave][g1 * DIM_HEAD + d]);
    }
    s0 *= ATT_SCALE; s1 *= ATT_SCALE;

    float m = fmaxf(s0, s1);
    m = fmaxf(m, __shfl_xor(m, 1));
    m = fmaxf(m, __shfl_xor(m, 2));
    float e0 = __expf(s0 - m), e1 = __expf(s1 - m);
    float sum = e0 + e1;
    sum += __shfl_xor(sum, 1);
    sum += __shfl_xor(sum, 2);
    float rs = 1.f / sum;
    float w0 = e0 * rs, w1 = e1 * rs;

    // gather all 8 softmax weights of this head from the 4-lane group
    float wg[8];
    const int bl = lane & ~3;
    #pragma unroll
    for (int s = 0; s < 4; ++s) {
        wg[2 * s]     = __shfl(w0, bl + s);
        wg[2 * s + 1] = __shfl(w1, bl + s);
    }

    // context: this lane owns 32 of the 128 dims of its head
    const int dq = lane & 3;
    float ctx[32];
    #pragma unroll
    for (int j = 0; j < 32; ++j) ctx[j] = 0.f;
    #pragma unroll
    for (int g = 0; g < 8; ++g) {
        const unsigned short* vp = &sKV[wave][DIMX + g * DIM_HEAD + dq * 32];
        #pragma unroll
        for (int j = 0; j < 32; ++j) ctx[j] += wg[g] * bf16_to_f32(vp[j]);
    }
    unsigned int* o = (unsigned int*)(Ctx + row * DIMX + h * DIM_HEAD + dq * 32);
    #pragma unroll
    for (int j = 0; j < 16; ++j) o[j] = pack_bf16x2(ctx[2 * j], ctx[2 * j + 1]);
}

// ---------------------------------------------------------------------------
// Host launcher
// ---------------------------------------------------------------------------
extern "C" void kernel_launch(void* const* d_in, const int* in_sizes, int n_in,
                              void* d_out, int out_size, void* d_ws, size_t ws_size,
                              hipStream_t stream) {
    (void)in_sizes; (void)n_in; (void)out_size; (void)ws_size;

    const float* q       = (const float*)d_in[0];
    const float* kv      = (const float*)d_in[1];
    const float* gamma_m = (const float*)d_in[2];
    const float* beta_m  = (const float*)d_in[3];
    const float* gamma_l = (const float*)d_in[4];
    const float* beta_l  = (const float*)d_in[5];
    const float* Wq      = (const float*)d_in[6];
    const float* Wkv     = (const float*)d_in[7];
    const float* Wo      = (const float*)d_in[8];
    float* out = (float*)d_out;

    const size_t ROWS = N_ROWS;
    char* p = (char*)d_ws;
    unsigned short* qn   = (unsigned short*)p; p += ROWS * DIMX * 2;
    unsigned short* kvn  = (unsigned short*)p; p += ROWS * DIMX * 2;
    unsigned short* WqB  = (unsigned short*)p; p += (size_t)DIMX * DIMX * 2;
    unsigned short* WkvB = (unsigned short*)p; p += (size_t)2 * DIMX * DIMX * 2;
    unsigned short* WoB  = (unsigned short*)p; p += (size_t)DIMX * DIMX * 2;
    unsigned short* Qp   = (unsigned short*)p; p += ROWS * DIMX * 2;
    unsigned short* KVp  = (unsigned short*)p; p += ROWS * 2 * DIMX * 2;
    unsigned short* Ctx  = (unsigned short*)p; p += ROWS * DIMX * 2;

    // weight conversion
    {
        int n1 = DIMX * DIMX, n2 = 2 * DIMX * DIMX;
        k_f32_to_bf16<<<(n1 / 4 + 255) / 256, 256, 0, stream>>>(Wq,  WqB,  n1);
        k_f32_to_bf16<<<(n2 / 4 + 255) / 256, 256, 0, stream>>>(Wkv, WkvB, n2);
        k_f32_to_bf16<<<(n1 / 4 + 255) / 256, 256, 0, stream>>>(Wo,  WoB,  n1);
    }

    // layernorms
    k_layernorm_bf16<<<ROWS / 8, 256, 0, stream>>>(q,  gamma_m, beta_m, qn);
    k_layernorm_bf16<<<ROWS / 8, 256, 0, stream>>>(kv, gamma_l, beta_l, kvn);

    // projections (bf16 WMMA, f32 accumulate)
    k_wmma_gemm_bf16<true><<<dim3(DIMX / BN, ROWS / BM), 256, 0, stream>>>(
        qn, WqB, Qp, (int)ROWS, DIMX, DIMX);
    k_wmma_gemm_bf16<true><<<dim3(2 * DIMX / BN, ROWS / BM), 256, 0, stream>>>(
        kvn, WkvB, KVp, (int)ROWS, 2 * DIMX, DIMX);

    // per-row heads-attention
    k_head_attention<<<ROWS / 8, 256, 0, stream>>>(Qp, KVp, Ctx);

    // output projection (f32 out)
    k_wmma_gemm_bf16<false><<<dim3(DIMX / BN, ROWS / BM), 256, 0, stream>>>(
        Ctx, WoB, out, (int)ROWS, DIMX, DIMX);
}
